// GridTransformerEncoder_39436389711919
// MI455X (gfx1250) — matrix operand order repaired and checked by reading, not compiled, not run on hardware
//
#include <hip/hip_runtime.h>

// ---------------------------------------------------------------------------
// GridTransformerEncoder for MI455X (gfx1250, wave32, WMMA).
// - All dense projections via v_wmma_f32_16x16x32_bf16 (bf16 in, f32 acc).
// - Weights pre-transposed to [N x K] so both A and B fragments load as
//   contiguous 128-bit chunks (global_load_b128 instead of 16x u16).
// - Flash self-attention: score WMMAs overlap an async V-tile copy to LDS
//   (global_load_async_to_lds_b128 + s_wait_asynccnt), P transposed
//   C->A layout through LDS, PV WMMAs with B fragments gathered from LDS.
// - Cross-attention (K=9 gathered neighbours) is scalar VALU; kv computed
//   once per grid point (LN+matmul are row-wise, commute with the gather).
// Working set fits in 192MB L2 -> compute bound -> bf16 WMMA throughput path.
// ---------------------------------------------------------------------------

typedef __bf16  v16bf __attribute__((ext_vector_type(16)));
typedef __bf16  v8bf  __attribute__((ext_vector_type(8)));
typedef float   v8f   __attribute__((ext_vector_type(8)));

static constexpr int LL  = 2;
static constexpr int DD  = 256;
static constexpr int HH  = 8;
static constexpr int DHH = 32;
static constexpr int FF  = 512;
static constexpr int KK  = 9;
static constexpr int MM  = 4;
static constexpr int GG  = 40;
static constexpr int NC  = GG * GG;   // 1600 grid tokens
static constexpr int NTT = 2048;      // target tokens

__device__ __forceinline__ float gelu_tanh(float x) {
    const float c0 = 0.7978845608028654f;   // sqrt(2/pi)
    const float c1 = 0.044715f;
    float x3 = x * x * x;
    return 0.5f * x * (1.0f + tanhf(c0 * (x + c1 * x3)));
}

// ---------------------------------------------------------------------------
// Generic bf16 GEMM: C[Mr x Nc] = A[Mr x Kd] * Bt^T (+bias)(gelu)(+Cin)
// A row-major [Mr x Kd]; Bt TRANSPOSED weights [Nc x Kd] so the B fragment
// (lane = column n, elements = K-sequential) loads as two b128 per step.
// One wave per 16x16 output tile (CDNA5 wave32 VGPR layouts, ISA 7.12.2).
// ---------------------------------------------------------------------------
template <bool ADD_C, bool BIAS, bool GELU_, bool OUT_BF16>
__global__ __launch_bounds__(256) void gemm_bf16_wmma(
    const __bf16* __restrict__ A, const __bf16* __restrict__ Bt,
    const float*  __restrict__ bias, const float* __restrict__ Cin,
    float* __restrict__ Cout, __bf16* __restrict__ CoutBf,
    int Mr, int Nc, int Kd)
{
    const int lane = threadIdx.x & 31;
    const int wave = blockIdx.x * (blockDim.x >> 5) + (threadIdx.x >> 5);
    const int ntiles = Nc >> 4;
    const int mtiles = Mr >> 4;
    const int tm = wave / ntiles;
    const int tn = wave - tm * ntiles;
    if (tm >= mtiles) return;               // wave-uniform exit, EXEC stays full

    const int mrow = lane & 15;
    const int g    = lane >> 4;             // 0: lanes 0-15, 1: lanes 16-31

    const __bf16* Arow = A  + (size_t)(tm * 16 + mrow) * Kd;
    const __bf16* Brow = Bt + (size_t)(tn * 16 + (lane & 15)) * Kd;

    v8f acc = {};
    for (int ko = 0; ko < Kd; ko += 32) {
        // A 16x32 fragment: lanes<16 hold K {0..7,16..23}; lanes>=16 {8..15,24..31}
        v8bf a0 = *(const v8bf*)(Arow + ko + g * 8);
        v8bf a1 = *(const v8bf*)(Arow + ko + 16 + g * 8);
        // B 32x16 fragment: lane holds column n; lanes<16 K {0..7,16..23}; lanes>=16 {8..15,24..31}
        v8bf b0 = *(const v8bf*)(Brow + ko + g * 8);
        v8bf b1 = *(const v8bf*)(Brow + ko + 16 + g * 8);
        __builtin_prefetch(Arow + ko + 32, 0, 1);    // global_prefetch_b8
        __builtin_prefetch(Brow + ko + 32, 0, 1);
        v16bf af, bfv;
        #pragma unroll
        for (int e = 0; e < 8; ++e) { af[e] = a0[e]; af[8 + e] = a1[e];
                                      bfv[e] = b0[e]; bfv[8 + e] = b1[e]; }
        acc = __builtin_amdgcn_wmma_f32_16x16x32_bf16(
                  false, af, false, bfv, (short)0, acc, false, false);
    }

    const int col = tn * 16 + (lane & 15);
    #pragma unroll
    for (int r = 0; r < 8; ++r) {
        const int row = tm * 16 + r + 8 * g;
        float v = acc[r];
        if (BIAS)  v += bias[col];
        if (GELU_) v  = gelu_tanh(v);
        if (ADD_C) v += Cin[(size_t)row * Nc + col];
        if (OUT_BF16) CoutBf[(size_t)row * Nc + col] = (__bf16)v;
        else          Cout  [(size_t)row * Nc + col] = v;
    }
}

// ---------------------------------------------------------------------------
// Flash self-attention over the 1600 grid tokens. One wave handles 16 queries
// of one (batch, head); keys stream in chunks of 32 with online softmax.
// The V tile (32 keys x 32 dh, 2KB) is copied to LDS asynchronously
// (ASYNCcnt) while the score WMMAs and softmax run, then gathered as the
// PV B fragment from LDS. qkv layout: [M*NC, 3*DD] (q | k | v).
// ---------------------------------------------------------------------------
__global__ __launch_bounds__(32) void self_attn_flash(
    const __bf16* __restrict__ qkv, __bf16* __restrict__ outb)
{
    __shared__ __bf16 Pst[16 * 32];     // P tile (C-layout -> A-layout staging)
    __shared__ __bf16 Vst[32 * 32];     // async-staged V tile [key][dh]

    const int lane = threadIdx.x;
    int bid = blockIdx.x;
    const int qt = bid % (NC / 16); bid /= (NC / 16);
    const int h  = bid % HH;        bid /= HH;
    const int b  = bid;

    const int mrow = lane & 15;
    const int g    = lane >> 4;
    const size_t rowbase = (size_t)b * NC;
    const int LDQ = 3 * DD;

    // Q fragment (A-layout, K = dh = 32), loaded once as 2x b128.
    const __bf16* Qrow = qkv + (rowbase + qt * 16 + mrow) * LDQ + h * DHH;
    v8bf q0 = *(const v8bf*)(Qrow + g * 16);
    v8bf q1 = *(const v8bf*)(Qrow + g * 16 + 8);
    v16bf aq;
    #pragma unroll
    for (int e = 0; e < 8; ++e) { aq[e] = q0[e]; aq[8 + e] = q1[e]; }

    // Per-lane addressing for the async V copy: each of 4 issues moves 16B/lane.
    const unsigned vst_base = (unsigned)(uintptr_t)(void*)&Vst[0] + (unsigned)lane * 16u;
    const int vrow  = lane >> 2;          // key row 0..7 (advances by 8 per issue)
    const int vcolB = (lane & 3) * 16;    // byte column within the 64B dh row

    v8f o0 = {}, o1 = {};
    float mrw[8], lsum[8];
    #pragma unroll
    for (int r = 0; r < 8; ++r) { mrw[r] = -3.0e30f; lsum[r] = 0.0f; }

    const float scale = 0.17677669529663687f;   // 1/sqrt(32)

    for (int kt = 0; kt < NC / 32; ++kt) {
        // ---- kick off async V-tile copy (2KB) into LDS ----
        {
            const char* vtile = (const char*)(qkv + (rowbase + kt * 32) * LDQ + 2 * DD + h * DHH);
            unsigned long long ga =
                (unsigned long long)(vtile + (size_t)vrow * (LDQ * 2) + vcolB);
            #pragma unroll
            for (int j = 0; j < 4; ++j) {
                unsigned           lo = vst_base + (unsigned)j * 512u;
                unsigned long long gj = ga + (unsigned long long)j * (8ull * LDQ * 2ull);
                asm volatile("global_load_async_to_lds_b128 %0, %1, off"
                             :: "v"(lo), "v"(gj) : "memory");
            }
        }
        // ---- scores: two 16q x 16k WMMAs (K = dh = 32), overlapping the copy ----
        v8f s0 = {}, s1 = {};
        {
            const __bf16* Kp0 = qkv + (rowbase + kt * 32 + (lane & 15)) * LDQ
                                + DD + h * DHH + g * 16;
            const __bf16* Kp1 = Kp0 + (size_t)16 * LDQ;
            v8bf k0a = *(const v8bf*)Kp0, k0b = *(const v8bf*)(Kp0 + 8);
            v8bf k1a = *(const v8bf*)Kp1, k1b = *(const v8bf*)(Kp1 + 8);
            v16bf b0, b1;
            #pragma unroll
            for (int e = 0; e < 8; ++e) { b0[e] = k0a[e]; b0[8 + e] = k0b[e];
                                          b1[e] = k1a[e]; b1[8 + e] = k1b[e]; }
            s0 = __builtin_amdgcn_wmma_f32_16x16x32_bf16(false, aq, false, b0, (short)0, s0, false, false);
            s1 = __builtin_amdgcn_wmma_f32_16x16x32_bf16(false, aq, false, b1, (short)0, s1, false, false);
        }
        // ---- online softmax per query row (reduce across the 16 N-lanes) ----
        #pragma unroll
        for (int r = 0; r < 8; ++r) {
            float v0 = s0[r] * scale, v1 = s1[r] * scale;
            float tmax = fmaxf(v0, v1);
            #pragma unroll
            for (int d = 1; d < 16; d <<= 1) tmax = fmaxf(tmax, __shfl_xor(tmax, d, 32));
            float nm    = fmaxf(mrw[r], tmax);
            float alpha = __expf(mrw[r] - nm);
            float p0 = __expf(v0 - nm), p1 = __expf(v1 - nm);
            float psum = p0 + p1;
            #pragma unroll
            for (int d = 1; d < 16; d <<= 1) psum += __shfl_xor(psum, d, 32);
            lsum[r] = lsum[r] * alpha + psum;
            o0[r] *= alpha;  o1[r] *= alpha;
            mrw[r] = nm;
            Pst[(r + 8 * g) * 32 + (lane & 15)]      = (__bf16)p0;
            Pst[(r + 8 * g) * 32 + 16 + (lane & 15)] = (__bf16)p1;
        }
        // ---- reload P as A fragment (16 queries x 32 keys) ----
        v16bf ap;
        #pragma unroll
        for (int e = 0; e < 8; ++e) ap[e]     = Pst[mrow * 32 + g * 16 + e];
        #pragma unroll
        for (int e = 0; e < 8; ++e) ap[8 + e] = Pst[mrow * 32 + g * 16 + 8 + e];
        // ---- wait for async V copy, gather PV B fragments from LDS ----
        asm volatile("s_wait_asynccnt 0x0" ::: "memory");
        v16bf bv0, bv1;
        #pragma unroll
        for (int e = 0; e < 16; ++e) bv0[e] = Vst[(g * 16 + e) * 32 + (lane & 15)];
        #pragma unroll
        for (int e = 0; e < 16; ++e) bv1[e] = Vst[(g * 16 + e) * 32 + 16 + (lane & 15)];
        o0 = __builtin_amdgcn_wmma_f32_16x16x32_bf16(false, ap, false, bv0, (short)0, o0, false, false);
        o1 = __builtin_amdgcn_wmma_f32_16x16x32_bf16(false, ap, false, bv1, (short)0, o1, false, false);
    }

    #pragma unroll
    for (int r = 0; r < 8; ++r) {
        const float inv = 1.0f / lsum[r];
        const int   row = qt * 16 + r + 8 * g;
        const size_t base = (rowbase + row) * DD + h * DHH;
        outb[base + (lane & 15)]      = (__bf16)(o0[r] * inv);
        outb[base + 16 + (lane & 15)] = (__bf16)(o1[r] * inv);
    }
}

// ---------------------------------------------------------------------------
// Cross-attention: one thread per (b, t, h). K=9 gathered neighbours, masked
// softmax. kvb is [M*NC, 2*DD] (per grid point, not per target-neighbour).
// ---------------------------------------------------------------------------
__global__ __launch_bounds__(256) void cross_attn_small(
    const __bf16* __restrict__ qb, const __bf16* __restrict__ kvb,
    const int* __restrict__ idx, const int* __restrict__ msk,
    __bf16* __restrict__ outb)
{
    int tid = blockIdx.x * blockDim.x + threadIdx.x;
    if (tid >= MM * NTT * HH) return;
    const int h = tid % HH;
    const int t = (tid / HH) % NTT;
    const int b = tid / (HH * NTT);
    const float scale = 0.17677669529663687f;

    float q[DHH];
    {
        const v8bf* qp = (const v8bf*)(qb + ((size_t)(b * NTT + t)) * DD + h * DHH);
        #pragma unroll
        for (int j = 0; j < 4; ++j) {
            v8bf v = qp[j];
            #pragma unroll
            for (int e = 0; e < 8; ++e) q[j * 8 + e] = (float)v[e];
        }
    }

    float s[KK];
    int   gi[KK];
    float smax = -3.0e30f;
    #pragma unroll
    for (int k = 0; k < KK; ++k) {
        const int gidx = idx[((size_t)(b * NTT + t)) * KK + k];
        gi[k] = gidx;
        const v8bf* kp = (const v8bf*)(kvb + ((size_t)(b * NC + gidx)) * (2 * DD) + h * DHH);
        float acc = 0.0f;
        #pragma unroll
        for (int j = 0; j < 4; ++j) {
            v8bf v = kp[j];
            #pragma unroll
            for (int e = 0; e < 8; ++e) acc += q[j * 8 + e] * (float)v[e];
        }
        const int vld = msk[((size_t)(b * NTT + t)) * KK + k];
        s[k] = vld ? acc * scale : -3.0e30f;
        smax = fmaxf(smax, s[k]);
    }
    float ssum = 0.0f;
    #pragma unroll
    for (int k = 0; k < KK; ++k) { s[k] = __expf(s[k] - smax); ssum += s[k]; }
    const float inv = 1.0f / ssum;

    float o[DHH];
    #pragma unroll
    for (int d = 0; d < DHH; ++d) o[d] = 0.0f;
    #pragma unroll
    for (int k = 0; k < KK; ++k) {
        const v8bf* vp = (const v8bf*)(kvb + ((size_t)(b * NC + gi[k])) * (2 * DD) + DD + h * DHH);
        const float w = s[k] * inv;
        #pragma unroll
        for (int j = 0; j < 4; ++j) {
            v8bf v = vp[j];
            #pragma unroll
            for (int e = 0; e < 8; ++e) o[j * 8 + e] += w * (float)v[e];
        }
    }
    __bf16* op = outb + ((size_t)(b * NTT + t)) * DD + h * DHH;
    #pragma unroll
    for (int d = 0; d < DHH; ++d) op[d] = (__bf16)o[d];
}

// ---------------------------------------------------------------------------
// LayerNorm over D=256 + cast to bf16. One 256-thread block per row.
// ---------------------------------------------------------------------------
__global__ __launch_bounds__(256) void ln_cast_bf16(
    const float* __restrict__ x, const float* __restrict__ gamma,
    const float* __restrict__ beta, __bf16* __restrict__ out)
{
    __shared__ float s1[256], s2[256];
    const int row = blockIdx.x;
    const int tid = threadIdx.x;
    const float v = x[(size_t)row * DD + tid];
    s1[tid] = v; s2[tid] = v * v;
    __syncthreads();
    for (int off = 128; off > 0; off >>= 1) {
        if (tid < off) { s1[tid] += s1[tid + off]; s2[tid] += s2[tid + off]; }
        __syncthreads();
    }
    const float mu  = s1[0] * (1.0f / DD);
    const float var = s2[0] * (1.0f / DD) - mu * mu;
    const float rs  = rsqrtf(var + 1e-5f);
    out[(size_t)row * DD + tid] = (__bf16)((v - mu) * rs * gamma[tid] + beta[tid]);
}

// ---------------------------------------------------------------------------
// Neighbour indices: per-dim argmin over the 40 axis values, 3x3 window with
// border clamping + validity mask (matches nearest_gridded_neighbours).
// ---------------------------------------------------------------------------
__global__ void build_idx_kernel(const float* __restrict__ xc,
                                 const float* __restrict__ xt,
                                 int* __restrict__ idx, int* __restrict__ msk)
{
    const int tid = blockIdx.x * blockDim.x + threadIdx.x;
    if (tid >= MM * NTT) return;
    const int b = tid / NTT;
    const float x0 = xt[(size_t)tid * 2 + 0];
    const float x1 = xt[(size_t)tid * 2 + 1];
    int n0 = 0, n1 = 0; float b0 = 3.4e38f, b1 = 3.4e38f;
    for (int i = 0; i < GG; ++i) {
        const float a0 = xc[(((size_t)b * GG + i) * GG + 0) * 2 + 0];
        const float d0 = fabsf(x0 - a0);
        if (d0 < b0) { b0 = d0; n0 = i; }
        const float a1 = xc[(((size_t)b * GG + 0) * GG + i) * 2 + 1];
        const float d1 = fabsf(x1 - a1);
        if (d1 < b1) { b1 = d1; n1 = i; }
    }
    #pragma unroll
    for (int kk = 0; kk < KK; ++kk) {
        const int d0 = kk / 3 - 1, d1 = kk % 3 - 1;
        const int r0 = n0 + d0, r1 = n1 + d1;
        const int vld = (r0 >= 0 && r0 < GG && r1 >= 0 && r1 < GG) ? 1 : 0;
        const int i0 = min(max(r0, 0), GG - 1);
        const int i1 = min(max(r1, 0), GG - 1);
        idx[(size_t)tid * KK + kk] = i0 * GG + i1;
        msk[(size_t)tid * KK + kk] = vld;
    }
}

// Cast f32 weights [K x N] row-major -> bf16 TRANSPOSED [N x K].
__global__ void cast_transpose_f32_bf16(const float* __restrict__ s,
                                        __bf16* __restrict__ d, int K, int N)
{
    const int i = blockIdx.x * blockDim.x + threadIdx.x;
    if (i >= K * N) return;
    const int n = i % N, k = i / N;           // coalesced read
    d[(size_t)n * K + k] = (__bf16)s[i];
}

// ---------------------------------------------------------------------------
// Host driver
// ---------------------------------------------------------------------------
static void launch_gemm(int variant, const __bf16* A, const __bf16* Bt,
                        const float* bias, const float* Cin,
                        float* Cout, __bf16* CoutBf,
                        int Mr, int Nc, int Kd, hipStream_t s)
{
    const int tiles   = (Mr / 16) * (Nc / 16);
    const int threads = tiles * 32;
    dim3 grid((threads + 255) / 256), block(256);
    switch (variant) {
    case 0: gemm_bf16_wmma<false, false, false, true ><<<grid, block, 0, s>>>(A, Bt, bias, Cin, Cout, CoutBf, Mr, Nc, Kd); break;
    case 1: gemm_bf16_wmma<true,  false, false, false><<<grid, block, 0, s>>>(A, Bt, bias, Cin, Cout, CoutBf, Mr, Nc, Kd); break;
    case 2: gemm_bf16_wmma<false, true,  true,  true ><<<grid, block, 0, s>>>(A, Bt, bias, Cin, Cout, CoutBf, Mr, Nc, Kd); break;
    case 3: gemm_bf16_wmma<true,  true,  false, false><<<grid, block, 0, s>>>(A, Bt, bias, Cin, Cout, CoutBf, Mr, Nc, Kd); break;
    }
}

extern "C" void kernel_launch(void* const* d_in, const int* in_sizes, int n_in,
                              void* d_out, int out_size, void* d_ws, size_t ws_size,
                              hipStream_t stream)
{
    (void)in_sizes; (void)n_in; (void)out_size; (void)ws_size;

    const float* xc      = (const float*)d_in[0];
    const float* zc_in   = (const float*)d_in[1];
    const float* xt      = (const float*)d_in[2];
    const float* zt_in   = (const float*)d_in[3];
    const float* sa_wqkv = (const float*)d_in[4];
    const float* sa_wo   = (const float*)d_in[5];
    const float* sa_ln1  = (const float*)d_in[6];
    const float* sa_ln2  = (const float*)d_in[7];
    const float* sa_w1   = (const float*)d_in[8];
    const float* sa_b1   = (const float*)d_in[9];
    const float* sa_w2   = (const float*)d_in[10];
    const float* sa_b2   = (const float*)d_in[11];
    const float* ca_wq   = (const float*)d_in[12];
    const float* ca_wkv  = (const float*)d_in[13];
    const float* ca_wo   = (const float*)d_in[14];
    const float* ca_lnq  = (const float*)d_in[15];
    const float* ca_lnkv = (const float*)d_in[16];
    const float* ca_ln2  = (const float*)d_in[17];
    const float* ca_w1   = (const float*)d_in[18];
    const float* ca_b1   = (const float*)d_in[19];
    const float* ca_w2   = (const float*)d_in[20];
    const float* ca_b2   = (const float*)d_in[21];

    // ---- workspace bump allocator ----
    char* wp = (char*)d_ws;
    auto alloc = [&](size_t bytes) -> void* {
        void* r = (void*)wp;
        wp += (bytes + 255) & ~(size_t)255;
        return r;
    };
    float*  zc      = (float*) alloc(sizeof(float)  * MM * NC * DD);
    __bf16* wqkv_bf = (__bf16*)alloc(sizeof(__bf16) * LL * DD * 3 * DD);
    __bf16* wo_bf   = (__bf16*)alloc(sizeof(__bf16) * LL * DD * DD);
    __bf16* w1_bf   = (__bf16*)alloc(sizeof(__bf16) * LL * DD * FF);
    __bf16* w2_bf   = (__bf16*)alloc(sizeof(__bf16) * LL * FF * DD);
    __bf16* cwq_bf  = (__bf16*)alloc(sizeof(__bf16) * LL * DD * DD);
    __bf16* cwkv_bf = (__bf16*)alloc(sizeof(__bf16) * LL * DD * 2 * DD);
    __bf16* cwo_bf  = (__bf16*)alloc(sizeof(__bf16) * LL * DD * DD);
    __bf16* cw1_bf  = (__bf16*)alloc(sizeof(__bf16) * LL * DD * FF);
    __bf16* cw2_bf  = (__bf16*)alloc(sizeof(__bf16) * LL * FF * DD);
    __bf16* hb      = (__bf16*)alloc(sizeof(__bf16) * MM * NTT * DD);     // LN outputs
    __bf16* hid     = (__bf16*)alloc(sizeof(__bf16) * MM * NTT * FF);     // FFN hidden
    __bf16* qkvb    = (__bf16*)alloc(sizeof(__bf16) * MM * NC * 3 * DD);  // self qkv
    __bf16* aout    = (__bf16*)alloc(sizeof(__bf16) * MM * NTT * DD);     // attn out
    __bf16* qb      = (__bf16*)alloc(sizeof(__bf16) * MM * NTT * DD);     // cross q
    __bf16* kvb     = (__bf16*)alloc(sizeof(__bf16) * MM * NC * 2 * DD);  // cross kv
    int*    idxb    = (int*)   alloc(sizeof(int)    * MM * NTT * KK);
    int*    mskb    = (int*)   alloc(sizeof(int)    * MM * NTT * KK);

    float* zt = (float*)d_out;  // zt accumulates in the output buffer

    hipMemcpyAsync(zc, zc_in, sizeof(float) * MM * NC * DD,  hipMemcpyDeviceToDevice, stream);
    hipMemcpyAsync(zt, zt_in, sizeof(float) * MM * NTT * DD, hipMemcpyDeviceToDevice, stream);

    // transpose-cast all weights to bf16 [N x K]
    auto castT = [&](const float* s, __bf16* d, int K, int N) {
        for (int l = 0; l < LL; ++l)
            cast_transpose_f32_bf16<<<(K * N + 255) / 256, 256, 0, stream>>>(
                s + (size_t)l * K * N, d + (size_t)l * K * N, K, N);
    };
    castT(sa_wqkv, wqkv_bf, DD, 3 * DD);
    castT(sa_wo,   wo_bf,   DD, DD);
    castT(sa_w1,   w1_bf,   DD, FF);
    castT(sa_w2,   w2_bf,   FF, DD);
    castT(ca_wq,   cwq_bf,  DD, DD);
    castT(ca_wkv,  cwkv_bf, DD, 2 * DD);
    castT(ca_wo,   cwo_bf,  DD, DD);
    castT(ca_w1,   cw1_bf,  DD, FF);
    castT(ca_w2,   cw2_bf,  FF, DD);

    build_idx_kernel<<<(MM * NTT + 255) / 256, 256, 0, stream>>>(xc, xt, idxb, mskb);

    const int RC = MM * NC;    // 6400 grid rows
    const int RT = MM * NTT;   // 8192 target rows

    for (int l = 0; l < LL; ++l) {
        // ================= self-attention on zc =================
        ln_cast_bf16<<<RC, 256, 0, stream>>>(zc, sa_ln1 + (size_t)l * 2 * DD,
                                             sa_ln1 + (size_t)l * 2 * DD + DD, hb);
        launch_gemm(0, hb, wqkv_bf + (size_t)l * DD * 3 * DD, nullptr, nullptr,
                    nullptr, qkvb, RC, 3 * DD, DD, stream);
        self_attn_flash<<<MM * HH * (NC / 16), 32, 0, stream>>>(qkvb, aout);
        launch_gemm(1, aout, wo_bf + (size_t)l * DD * DD, nullptr, zc,
                    zc, nullptr, RC, DD, DD, stream);
        // ================= FFN on zc =================
        ln_cast_bf16<<<RC, 256, 0, stream>>>(zc, sa_ln2 + (size_t)l * 2 * DD,
                                             sa_ln2 + (size_t)l * 2 * DD + DD, hb);
        launch_gemm(2, hb, w1_bf + (size_t)l * DD * FF, sa_b1 + (size_t)l * FF,
                    nullptr, nullptr, hid, RC, FF, DD, stream);
        launch_gemm(3, hid, w2_bf + (size_t)l * FF * DD, sa_b2 + (size_t)l * DD,
                    zc, zc, nullptr, RC, DD, FF, stream);
        // ================= cross-attention zt <- zc =================
        ln_cast_bf16<<<RT, 256, 0, stream>>>(zt, ca_lnq + (size_t)l * 2 * DD,
                                             ca_lnq + (size_t)l * 2 * DD + DD, hb);
        launch_gemm(0, hb, cwq_bf + (size_t)l * DD * DD, nullptr, nullptr,
                    nullptr, qb, RT, DD, DD, stream);
        ln_cast_bf16<<<RC, 256, 0, stream>>>(zc, ca_lnkv + (size_t)l * 2 * DD,
                                             ca_lnkv + (size_t)l * 2 * DD + DD, hb);
        launch_gemm(0, hb, cwkv_bf + (size_t)l * DD * 2 * DD, nullptr, nullptr,
                    nullptr, kvb, RC, 2 * DD, DD, stream);
        cross_attn_small<<<(MM * NTT * HH + 255) / 256, 256, 0, stream>>>(
            qb, kvb, idxb, mskb, aout);
        launch_gemm(1, aout, cwo_bf + (size_t)l * DD * DD, nullptr, zt,
                    zt, nullptr, RT, DD, DD, stream);
        // ================= FFN on zt =================
        ln_cast_bf16<<<RT, 256, 0, stream>>>(zt, ca_ln2 + (size_t)l * 2 * DD,
                                             ca_ln2 + (size_t)l * 2 * DD + DD, hb);
        launch_gemm(2, hb, cw1_bf + (size_t)l * DD * FF, ca_b1 + (size_t)l * FF,
                    nullptr, nullptr, hid, RT, FF, DD, stream);
        launch_gemm(3, hid, cw2_bf + (size_t)l * FF * DD, ca_b2 + (size_t)l * DD,
                    zt, zt, nullptr, RT, DD, FF, stream);
    }
}
